// Gemma3nTextAttention_73306501808590
// MI455X (gfx1250) — compile-verified, matrix operand order checked by reading, not run on hardware
//
#include <hip/hip_runtime.h>
#include <hip/hip_bf16.h>

// Problem constants (match reference).
#define B_ 2
#define S_ 2048
#define D_ 2048
#define H_ 8
#define KV_ 2
#define HD_ 256
#define WINDOW_ 512

typedef __attribute__((ext_vector_type(16))) _Float16 v16h;
typedef __attribute__((ext_vector_type(8)))  _Float16 v8h;
typedef __attribute__((ext_vector_type(8)))  float    v8f;

__device__ __forceinline__ v16h mk16(v8h lo, v8h hi) {
  return __builtin_shufflevector(lo, hi, 0,1,2,3,4,5,6,7,8,9,10,11,12,13,14,15);
}

__device__ __forceinline__ v8f wmma16(v16h a, v16h b, v8f c) {
  // D(f32 16x16) = A(16x32 f16) * B(32x16 f16) + C
  return __builtin_amdgcn_wmma_f32_16x16x32_f16(false, a, false, b, (short)0, c, false, false);
}

// f16-accumulator WMMA for the PV product: halves accumulator VGPR footprint.
#if defined(__has_builtin)
#if __has_builtin(__builtin_amdgcn_wmma_f16_16x16x32_f16)
#define HAVE_WMMA_F16C 1
#endif
#endif

#ifdef HAVE_WMMA_F16C
__device__ __forceinline__ v8h wmma16h(v16h a, v16h b, v8h c) {
  // D(f16 16x16) = A(16x32 f16) * B(32x16 f16) + C(f16)
  return __builtin_amdgcn_wmma_f16_16x16x32_f16(false, a, false, b, (short)0, c, false, false);
}
typedef v8h accv_t;
#else
typedef v8f accv_t;
#endif

// ---------------------------------------------------------------------------
// Tiled WMMA GEMM: C[M,N] = A[M,K] * Bw[K,N].  A is TA (float or _Float16),
// Bw is f32, C is f32.  Block = 256 threads (8 waves), block tile 128x64,
// wave tile 32x32 (2x2 WMMA C tiles) -> 4 WMMAs per wave per K-step of 32,
// staged through LDS with f32->fp16 conversion on the staging store.
// Requires M % 128 == 0, N % 64 == 0, K % 32 == 0 (true for all uses here).
// ---------------------------------------------------------------------------
template <typename TA>
__global__ __launch_bounds__(256) void gemm_wmma(const TA* __restrict__ A,
                                                 const float* __restrict__ Bw,
                                                 float* __restrict__ C,
                                                 int M, int N, int K) {
  __shared__ __align__(16) _Float16 As[128][40];  // 128 rows x 32 cols (+pad)
  __shared__ __align__(16) _Float16 Bs[32][72];   // 32 rows x 64 cols (+pad)
  const int tid  = threadIdx.x;
  const int lane = tid & 31;
  const int wave = tid >> 5;
  const int wm = wave & 3;          // M sub-tile: 32 rows each (4 waves)
  const int wn = wave >> 2;         // N sub-tile: 32 cols each (2 waves)
  const int bm = blockIdx.y * 128;
  const int bn = blockIdx.x * 64;
  const int g   = lane >> 4;        // lane group per ISA fragment layout
  const int n16 = lane & 15;

  v8f acc[2][2];
  #pragma unroll
  for (int mi = 0; mi < 2; ++mi)
    #pragma unroll
    for (int ni = 0; ni < 2; ++ni) acc[mi][ni] = (v8f){};

  for (int k0 = 0; k0 < K; k0 += 32) {
    { // stage A: each thread converts 16 consecutive elements of one row
      const int r = tid >> 1, c = (tid & 1) * 16;
      const TA* src = A + (size_t)(bm + r) * K + k0 + c;
      #pragma unroll
      for (int i = 0; i < 16; ++i) As[r][c + i] = (_Float16)src[i];
    }
    { // stage B: each thread converts 8 consecutive elements of one row
      const int r = tid >> 3, c = (tid & 7) * 8;
      const float* src = Bw + (size_t)(k0 + r) * N + bn + c;
      #pragma unroll
      for (int i = 0; i < 8; ++i) Bs[r][c + i] = (_Float16)src[i];
    }
    __syncthreads();

    // A fragments: lane row m, K runs [g*8, +8) and [16+g*8, +8)
    v16h af[2];
    #pragma unroll
    for (int mi = 0; mi < 2; ++mi) {
      const int arow = wm * 32 + mi * 16 + n16;
      af[mi] = mk16(*(const v8h*)&As[arow][g * 8],
                    *(const v8h*)&As[arow][16 + g * 8]);
    }
    // B fragments: lane col n, K rows per kmap
    v16h bf[2];
    #pragma unroll
    for (int e = 0; e < 16; ++e) {
      const int kr = (e < 8) ? (g * 8 + e) : (16 + g * 8 + (e - 8));
      bf[0][e] = Bs[kr][wn * 32 + n16];
      bf[1][e] = Bs[kr][wn * 32 + 16 + n16];
    }
    #pragma unroll
    for (int mi = 0; mi < 2; ++mi)
      #pragma unroll
      for (int ni = 0; ni < 2; ++ni)
        acc[mi][ni] = wmma16(af[mi], bf[ni], acc[mi][ni]);
    __syncthreads();
  }

  // C layout: VGPR r -> M = r + 8*g, N = n16
  #pragma unroll
  for (int mi = 0; mi < 2; ++mi)
    #pragma unroll
    for (int ni = 0; ni < 2; ++ni)
      #pragma unroll
      for (int r = 0; r < 8; ++r) {
        const int row = bm + wm * 32 + mi * 16 + r + 8 * g;
        const int col = bn + wn * 32 + ni * 16 + n16;
        C[(size_t)row * N + col] = acc[mi][ni][r];
      }
}

// ---------------------------------------------------------------------------
// RMSNorm + RoPE for Q/K.  One wave per (b,s,head) row of HD=256.
// Reads f32 lin [b*S+s][nheads*HD], writes fp16 head-major out[b][h][s][hd].
// ---------------------------------------------------------------------------
__global__ __launch_bounds__(256) void norm_rope(const float* __restrict__ lin,
                                                 const float* __restrict__ cosb,
                                                 const float* __restrict__ sinb,
                                                 const float* __restrict__ w,
                                                 _Float16* __restrict__ outp,
                                                 int nheads) {
  const int idx  = blockIdx.x * 8 + (threadIdx.x >> 5);
  const int lane = threadIdx.x & 31;
  const int h = idx % nheads;
  const int s = (idx / nheads) % S_;
  const int b = idx / (nheads * S_);
  const int d0 = lane * 8;

  const float* src = lin + ((size_t)(b * S_ + s) * nheads + h) * HD_;
  float x[8];
  float ss = 0.f;
  #pragma unroll
  for (int i = 0; i < 8; ++i) { x[i] = src[d0 + i]; ss += x[i] * x[i]; }
  #pragma unroll
  for (int m = 1; m < 32; m <<= 1) ss += __shfl_xor(ss, m, 32);
  const float rms = rsqrtf(ss * (1.0f / HD_) + 1e-6f);

  const float* cp = cosb + (size_t)(b * S_ + s) * HD_ + d0;
  const float* sp = sinb + (size_t)(b * S_ + s) * HD_ + d0;
  _Float16* dst = outp + (((size_t)b * nheads + h) * S_ + s) * HD_ + d0;

  float y[8];
  #pragma unroll
  for (int i = 0; i < 8; ++i) y[i] = x[i] * rms * w[d0 + i];
  #pragma unroll
  for (int i = 0; i < 8; ++i) {
    const float partner = __shfl_xor(y[i], 16, 32);      // d +/- 128
    const float rot = (lane < 16) ? -partner : partner;  // rot = [-x2, x1]
    dst[i] = (_Float16)(y[i] * cp[i] + rot * sp[i]);
  }
}

// ---------------------------------------------------------------------------
// RMSNorm for V (weight = 1), written TRANSPOSED: vt[b][kv][hd][s] (fp16),
// so PV B-fragments in attention are contiguous 16B loads.
// ---------------------------------------------------------------------------
__global__ __launch_bounds__(256) void v_norm(const float* __restrict__ lin,
                                              _Float16* __restrict__ vt) {
  const int idx  = blockIdx.x * 8 + (threadIdx.x >> 5);
  const int lane = threadIdx.x & 31;
  const int kv = idx % KV_;
  const int s  = (idx / KV_) % S_;
  const int b  = idx / (KV_ * S_);
  const int d0 = lane * 8;

  const float* src = lin + ((size_t)(b * S_ + s) * KV_ + kv) * HD_;
  float x[8];
  float ss = 0.f;
  #pragma unroll
  for (int i = 0; i < 8; ++i) { x[i] = src[d0 + i]; ss += x[i] * x[i]; }
  #pragma unroll
  for (int m = 1; m < 32; m <<= 1) ss += __shfl_xor(ss, m, 32);
  const float rms = rsqrtf(ss * (1.0f / HD_) + 1e-6f);

  _Float16* base = vt + ((size_t)(b * KV_ + kv)) * HD_ * S_;
  #pragma unroll
  for (int i = 0; i < 8; ++i)
    base[(size_t)(d0 + i) * S_ + s] = (_Float16)(x[i] * rms);
}

// ---------------------------------------------------------------------------
// Flash-style sliding-window attention.  One wave per 16-query tile of one
// (b,h); 32 keys per iteration.  Q fragments register-resident (64 VGPRs).
// PV accumulation uses V_WMMA_F16_16X16X32_F16 (f16 C/D) when available:
// 16 d-tiles x 4 VGPRs = 64 VGPRs instead of 128, which brings the wave
// under 256 VGPRs so the K/V double-buffers get distinct registers and
// global_load_b128 clauses overlap WMMA issue.  K/V (8 MB fp16) live in L2.
// ---------------------------------------------------------------------------
__global__ __launch_bounds__(128) void attn(const _Float16* __restrict__ qh,
                                            const _Float16* __restrict__ kh,
                                            const _Float16* __restrict__ vth,
                                            _Float16* __restrict__ ctx) {
  __shared__ __align__(16) _Float16 pbuf[4][16][32];   // per-wave P relayout
  const int wave = threadIdx.x >> 5;
  const int lane = threadIdx.x & 31;
  const int wg = blockIdx.x * 4 + wave;
  const int qt = wg & 127;            // S/16 = 128 query tiles
  const int h  = (wg >> 7) & 7;
  const int b  = wg >> 10;
  const int q0 = qt * 16;
  const int kv = h >> 2;              // H/KV = 4
  const int g   = lane >> 4;
  const int n16 = lane & 15;

  const _Float16* qp = qh  + ((size_t)b * H_  + h ) * S_  * HD_;
  const _Float16* kp = kh  + ((size_t)b * KV_ + kv) * S_  * HD_;
  const _Float16* vp = vth + ((size_t)b * KV_ + kv) * HD_ * S_;

  // Q fragments: A-layout, lane row m = n16, 8 chunks of K=32 over HD=256.
  v16h qfrag[8];
  #pragma unroll
  for (int c = 0; c < 8; ++c) {
    const _Float16* r = qp + (size_t)(q0 + n16) * HD_ + c * 32;
    qfrag[c] = mk16(*(const v8h*)(r + g * 8), *(const v8h*)(r + 16 + g * 8));
  }

  accv_t acc[16];
  #pragma unroll
  for (int dt = 0; dt < 16; ++dt) acc[dt] = (accv_t){};
  float mrun[8], lrun[8];
  #pragma unroll
  for (int r = 0; r < 8; ++r) { mrun[r] = -1e30f; lrun[r] = 0.f; }

  int tlo = q0 - (WINDOW_ - 1); if (tlo < 0) tlo = 0; tlo >>= 4;
  const int thi = (q0 + 15) >> 4;

  for (int kt = tlo; kt <= thi; kt += 2) {
    const int kbaseA = kt * 16;
    const int kbaseB = kbaseA + 16;
    const int keyA = min(kbaseA + n16, S_ - 1);   // clamp: masked anyway
    const int keyB = min(kbaseB + n16, S_ - 1);
    const _Float16* rowA = kp + (size_t)keyA * HD_;
    const _Float16* rowB = kp + (size_t)keyB * HD_;

    // scores for two 16-key tiles, K fragments register double-buffered
    v8f s0 = {}, s1 = {};
    v16h ka = mk16(*(const v8h*)(rowA + g * 8),
                   *(const v8h*)(rowA + 16 + g * 8));
    v16h kb = mk16(*(const v8h*)(rowB + g * 8),
                   *(const v8h*)(rowB + 16 + g * 8));
    #pragma unroll
    for (int c = 0; c < 8; ++c) {
      v16h kan = ka, kbn = kb;
      if (c < 7) {  // prefetch next chunk before this chunk's WMMAs
        const int o = (c + 1) * 32;
        kan = mk16(*(const v8h*)(rowA + o + g * 8),
                   *(const v8h*)(rowA + o + 16 + g * 8));
        kbn = mk16(*(const v8h*)(rowB + o + g * 8),
                   *(const v8h*)(rowB + o + 16 + g * 8));
      }
      s0 = wmma16(qfrag[c], ka, s0);
      s1 = wmma16(qfrag[c], kb, s1);
      ka = kan; kb = kbn;
    }

    // online softmax (row = q0 + r + 8*g, key col = n16).
    // (unsigned)(qrow-k) < WINDOW encodes causality AND the window together.
    float pA[8], pB[8], scl[8];
    #pragma unroll
    for (int r = 0; r < 8; ++r) {
      const int qrow = q0 + r + 8 * g;
      const float sa = s0[r] +
          (((unsigned)(qrow - (kbaseA + n16)) < WINDOW_) ? 0.f : -1e9f);
      const float sb = s1[r] +
          (((unsigned)(qrow - (kbaseB + n16)) < WINDOW_) ? 0.f : -1e9f);
      float mx = fmaxf(sa, sb);
      #pragma unroll
      for (int m = 1; m < 16; m <<= 1) mx = fmaxf(mx, __shfl_xor(mx, m, 16));
      const float mn = fmaxf(mrun[r], mx);
      scl[r] = __expf(mrun[r] - mn);
      pA[r] = __expf(sa - mn);
      pB[r] = __expf(sb - mn);
      float rs = pA[r] + pB[r];
      #pragma unroll
      for (int m = 1; m < 16; m <<= 1) rs += __shfl_xor(rs, m, 16);
      lrun[r] = lrun[r] * scl[r] + rs;
      mrun[r] = mn;
    }
    // rescale accumulators; f16 C/D rows map j -> M = j + 8*g, same as f32
    accv_t sv;
    #pragma unroll
    for (int r = 0; r < 8; ++r)
#ifdef HAVE_WMMA_F16C
      sv[r] = (_Float16)scl[r];
#else
      sv[r] = scl[r];
#endif
    #pragma unroll
    for (int dt = 0; dt < 16; ++dt) acc[dt] *= sv;

    // restage P (C layout) -> LDS -> A layout fragment (16 queries x 32 keys)
    #pragma unroll
    for (int r = 0; r < 8; ++r) {
      pbuf[wave][r + 8 * g][n16]      = (_Float16)pA[r];
      pbuf[wave][r + 8 * g][16 + n16] = (_Float16)pB[r];
    }
    v16h pf;
    #pragma unroll
    for (int e = 0; e < 16; ++e) {
      const int kk = (e < 8) ? (g * 8 + e) : (16 + g * 8 + (e - 8));
      pf[e] = pbuf[wave][n16][kk];
    }

    // PV: 16 d-tiles from transposed V, register double-buffered
    const int sb0 = min(kbaseA + g * 8, S_ - 8);
    const int sb1 = min(kbaseA + 16 + g * 8, S_ - 8);
    v16h vf = mk16(*(const v8h*)(vp + (size_t)n16 * S_ + sb0),
                   *(const v8h*)(vp + (size_t)n16 * S_ + sb1));
    #pragma unroll
    for (int dt = 0; dt < 16; ++dt) {
      v16h vn = vf;
      if (dt < 15) {
        const _Float16* vr = vp + (size_t)((dt + 1) * 16 + n16) * S_;
        vn = mk16(*(const v8h*)(vr + sb0), *(const v8h*)(vr + sb1));
      }
#ifdef HAVE_WMMA_F16C
      acc[dt] = wmma16h(pf, vf, acc[dt]);
#else
      acc[dt] = wmma16(pf, vf, acc[dt]);
#endif
      vf = vn;
    }
  }

  // epilogue: normalize and scatter fp16 ctx[b][s][h*HD + d]
  float inv[8];
  #pragma unroll
  for (int r = 0; r < 8; ++r) inv[r] = 1.f / lrun[r];
  #pragma unroll
  for (int dt = 0; dt < 16; ++dt)
    #pragma unroll
    for (int r = 0; r < 8; ++r) {
      const int row = q0 + r + 8 * g;
      const int col = h * HD_ + dt * 16 + n16;
      ctx[((size_t)b * S_ + row) * (H_ * HD_) + col] =
          (_Float16)((float)acc[dt][r] * inv[r]);
    }
}

// ---------------------------------------------------------------------------
// Host-side orchestration.  Workspace layout (bytes):
//   q_lin f32 [4096x2048]  @ 0          (32 MiB)
//   k_lin f32 [4096x512]   @ 32 MiB     ( 8 MiB)
//   v_lin f32 [4096x512]   @ 40 MiB     ( 8 MiB)
//   q_h  fp16 [B][H][S][HD]  @ 48 MiB   (16 MiB)
//   k_h  fp16 [B][KV][S][HD] @ 64 MiB   ( 4 MiB)
//   vt_h fp16 [B][KV][HD][S] @ 68 MiB   ( 4 MiB)
//   ctx  fp16 [B][S][H*HD]   @ 72 MiB   (16 MiB)   total 88 MiB
// ---------------------------------------------------------------------------
extern "C" void kernel_launch(void* const* d_in, const int* in_sizes, int n_in,
                              void* d_out, int out_size, void* d_ws, size_t ws_size,
                              hipStream_t stream) {
  const float* hs   = (const float*)d_in[0];
  const float* cosb = (const float*)d_in[1];
  const float* sinb = (const float*)d_in[2];
  // d_in[3] attention_mask: recomputed analytically in the attention kernel
  const float* wq = (const float*)d_in[4];
  const float* wk = (const float*)d_in[5];
  const float* wv = (const float*)d_in[6];
  const float* wo = (const float*)d_in[7];
  const float* qw = (const float*)d_in[8];
  const float* kw = (const float*)d_in[9];

  char* ws = (char*)d_ws;
  float*    q_lin = (float*)(ws + 0);
  float*    k_lin = (float*)(ws + (size_t)33554432);
  float*    v_lin = (float*)(ws + (size_t)41943040);
  _Float16* q_h   = (_Float16*)(ws + (size_t)50331648);
  _Float16* k_h   = (_Float16*)(ws + (size_t)67108864);
  _Float16* vt_h  = (_Float16*)(ws + (size_t)71303168);
  _Float16* ctx   = (_Float16*)(ws + (size_t)75497472);
  float* outp = (float*)d_out;

  const int M = B_ * S_;  // 4096

  // QKV projections
  gemm_wmma<float><<<dim3((H_ * HD_) / 64, M / 128), 256, 0, stream>>>(hs, wq, q_lin, M, H_ * HD_, D_);
  gemm_wmma<float><<<dim3((KV_ * HD_) / 64, M / 128), 256, 0, stream>>>(hs, wk, k_lin, M, KV_ * HD_, D_);
  gemm_wmma<float><<<dim3((KV_ * HD_) / 64, M / 128), 256, 0, stream>>>(hs, wv, v_lin, M, KV_ * HD_, D_);

  // Per-head RMSNorm (+RoPE for q/k), head-major fp16 relayout
  norm_rope<<<(B_ * S_ * H_) / 8, 256, 0, stream>>>(q_lin, cosb, sinb, qw, q_h, H_);
  norm_rope<<<(B_ * S_ * KV_) / 8, 256, 0, stream>>>(k_lin, cosb, sinb, kw, k_h, KV_);
  v_norm<<<(B_ * S_ * KV_) / 8, 256, 0, stream>>>(v_lin, vt_h);

  // Sliding-window attention: B*H*(S/16) = 2048 waves, 4 waves/block
  attn<<<(B_ * H_ * (S_ / 16)) / 4, 128, 0, stream>>>(q_h, k_h, vt_h, ctx);

  // Output projection straight into d_out
  gemm_wmma<_Float16><<<dim3(D_ / 64, M / 128), 256, 0, stream>>>(ctx, wo, outp, M, D_, H_ * HD_);
}
